// ViT_80272938762347
// MI455X (gfx1250) — matrix operand order repaired and checked by reading, not compiled, not run on hardware
//
#include <hip/hip_runtime.h>
#include <hip/hip_bf16.h>

// ---------------------------------------------------------------------------
// ViT encoder forward for MI455X (gfx1250, wave32, WMMA + TDM).
// One wave per image; 4 waves (128 threads) per workgroup.
// Dense projections: v_wmma_f32_16x16x32_f16 (f16 in, f32 acc).
// Attention core:    v_wmma_f32_16x16x4_f32  (fp32 throughout softmax path).
// Image ingest:      tensor_load_to_lds (TDM DMA, overlaps weight staging),
//                    drained with s_wait_tensorcnt; falls back to coalesced
//                    float4 global loads if the builtin is unavailable.
// Fragment traffic:  128-bit LDS loads (alignas + fragment-major weights).
// ---------------------------------------------------------------------------

typedef _Float16 v16h __attribute__((ext_vector_type(16)));
typedef _Float16 v8h  __attribute__((ext_vector_type(8)));
typedef _Float16 h2   __attribute__((ext_vector_type(2)));
typedef float    v8f  __attribute__((ext_vector_type(8)));
typedef float    v2f  __attribute__((ext_vector_type(2)));
typedef unsigned int v4u __attribute__((ext_vector_type(4)));
typedef int      v8i  __attribute__((ext_vector_type(8)));
typedef int      v4i  __attribute__((ext_vector_type(4)));

#if __has_builtin(__builtin_amdgcn_tensor_load_to_lds) && __has_builtin(__builtin_amdgcn_s_wait_tensorcnt)
#define USE_TDM 1
#else
#define USE_TDM 0
#endif

#define WAVES_PER_WG 4
#define THREADS      (WAVES_PER_WG * 32)

#define S_TOK 16
#define NHEAD 4
#define PDIM 49
#define MLPH 64
#define NCLS 10

union V16U { v16h v; v8h h[2]; };

// ---- WMMA fragment loaders (CDNA5 ISA 7.12.2) -------------------

// A: 16xK f16 slice, K-chunk of 32 at `ko`, row stride `rs` (elems, mult of 8).
// lane<16: M=lane, K = {ko+0..7, ko+16..23}; lane>=16: K = {ko+8..15, ko+24..31}
// Both 8-element groups are contiguous -> two ds_load_b128.
__device__ __forceinline__ v16h load_a16(const _Float16* src, int rs, int ko, int lane) {
    int m  = lane & 15;
    int kh = (lane >> 4) << 3;
    V16U u;
    u.h[0] = *(const v8h*)(src + m * rs + ko + kh);
    u.h[1] = *(const v8h*)(src + m * rs + ko + 16 + kh);
    return u.v;
}

// B operand from fragment-major staged weights: frag f, lane L hold 16
// contiguous f16 -> two ds_load_b128.
__device__ __forceinline__ v16h load_bfrag(const _Float16* Wf, int frag, int lane) {
    const v8h* p = (const v8h*)(Wf + ((frag << 5) + lane) * 16);
    V16U u;
    u.h[0] = p[0];
    u.h[1] = p[1];
    return u.v;
}

__device__ __forceinline__ v8f wmma_f16(v16h a, v16h b, v8f c) {
    return __builtin_amdgcn_wmma_f32_16x16x32_f16(false, a, false, b, (short)0, c, false, false);
}

// Stage a logical [K x N] f16 weight into fragment-major layout:
// dst[((tile*chunks + chunk)*32 + lane)*16 + e], where per B-layout
// K = chunk*32 + (e>=8 ? 16 : 0) + (lane>=16 ? 8 : 0) + (e&7), N = tile*16 + (lane&15).
template <typename F>
__device__ __forceinline__ void stage_frag(_Float16* dst, int tiles, int chunks,
                                           int tid, F getw) {
    int total = tiles * chunks * 512;
    for (int idx = tid; idx < total; idx += THREADS) {
        int f    = idx >> 9;
        int r    = idx & 511;
        int lane = r >> 4;
        int e    = r & 15;
        int tile  = f / chunks;
        int chunk = f - tile * chunks;
        int K = chunk * 32 + ((e >> 3) << 4) + ((lane >> 4) << 3) + (e & 7);
        int N = tile * 16 + (lane & 15);
        dst[idx] = (_Float16)getw(K, N);
    }
}

// Per-wave LDS working set (all 16B aligned for b128 LDS traffic).
// The raw-image DMA landing zone overlays q/k/v/o (dead until after embed).
struct WaveBuf {
    alignas(16) _Float16 ast[16 * 64];  // f16 A staging (stride 64)
    alignas(16) float    x[16 * 32];    // x, later mhsa (in-place residual)
    union {
        struct {
            alignas(16) _Float16 q[16 * 32];  // q; later relu(h2@W1) staging
            alignas(16) _Float16 k[16 * 32];
            alignas(16) _Float16 v[16 * 32];
            alignas(16) float    o[16 * 32];  // attention output
        };
        alignas(16) float raw[784];           // TDM destination (28x28 f32)
    } u;
    alignas(16) float sc[16 * 16];      // per-head scores / softmax
    alignas(16) float hid[64];
    alignas(16) float cls[32];
};

__global__ __launch_bounds__(THREADS) void vit_fwd_kernel(
    const float* __restrict__ img,
    const float* __restrict__ W_map, const float* __restrict__ b_map,
    const float* __restrict__ ln_g,  const float* __restrict__ ln_b,
    const float* __restrict__ Wq, const float* __restrict__ bq,
    const float* __restrict__ Wk, const float* __restrict__ bk,
    const float* __restrict__ Wv, const float* __restrict__ bv,
    const float* __restrict__ Wo, const float* __restrict__ bo,
    const float* __restrict__ W1, const float* __restrict__ b1,
    const float* __restrict__ W2, const float* __restrict__ b2,
    const float* __restrict__ Wc1, const float* __restrict__ bc1,
    const float* __restrict__ Wc2, const float* __restrict__ bc2,
    float* __restrict__ out)
{
    __shared__ WaveBuf wb[WAVES_PER_WG];
    // Fragment-major staged weights
    __shared__ alignas(32) _Float16 sWmapF[4 * 512];            // 2 N-tiles x 2 K-chunks
    __shared__ alignas(32) _Float16 sWqF[2 * 512], sWkF[2 * 512], sWvF[2 * 512], sWoF[2 * 512];
    __shared__ alignas(32) _Float16 sW1F[2 * 512], sW2F[2 * 512];
    __shared__ float sBmap[32], sBq[32], sBk[32], sBv[32], sBo[32], sB1[32], sB2[32], sLg[32], sLb[32];

    const int tid  = threadIdx.x;
    const int wave = tid >> 5;
    const int lane = tid & 31;
    const int imgIdx = blockIdx.x * WAVES_PER_WG + wave;

    WaveBuf& w = wb[wave];

#if USE_TDM
    // ---- Issue TDM DMA of this wave's image into LDS (overlaps weight
    //      staging below). 1D tensor: 784 x f32, tile = whole tensor.
    //      Descriptor per CDNA5 ISA.
    {
        unsigned lds_off = (unsigned)(uintptr_t)(void*)w.u.raw;   // flat LDS addr[31:0] = LDS byte offset
        unsigned long long ga =
            (unsigned long long)(uintptr_t)(img + (size_t)imgIdx * 784);
        v4u g0;
        g0.x = 1u;                                             // count=1 valid descriptor
        g0.y = lds_off;                                        // D#.lds_addr
        g0.z = (unsigned)ga;                                   // global_addr[31:0]
        g0.w = (unsigned)((ga >> 32) & 0x01FFFFFFu) | (2u << 30); // global_addr[56:32] | type=2
        v8i g1;
        g1[0] = 0x20000;                 // workgroup_mask=0, data_size=2 (4B)
        g1[1] = (int)(784u << 16);       // tensor_dim0[15:0]=784 in bits[31:16]
        g1[2] = (int)(1u << 16);         // tensor_dim0 hi=0; tensor_dim1=1
        g1[3] = (int)(784u << 16);       // tensor_dim1 hi=0; tile_dim0=784
        g1[4] = 1;                       // tile_dim1=1, tile_dim2=0
        g1[5] = 784;                     // tensor_dim0_stride[31:0]
        g1[6] = 0;                       // stride hi / dim1_stride lo
        g1[7] = 0;
        v4i z4 = { 0, 0, 0, 0 };         // groups 2/3 unused (<=2D tensor)
        v8i z8 = { 0, 0, 0, 0, 0, 0, 0, 0 };
        // 6-arg form on this toolchain (therock/clang-23 signature):
        // (v4u g0, v8i g1, v4i g2, v4i g3, v8i extra, i32 cpol)
        __builtin_amdgcn_tensor_load_to_lds(g0, g1, z4, z4, z8, 0);
    }
#endif

    // ---- Stage weights once per workgroup (f32 -> f16, fragment-major) ----
    stage_frag(sWmapF, 2, 2, tid, [&](int K, int N) {
        return (K < PDIM && N >= 8) ? W_map[K * 24 + (N - 8)] : 0.f;
    });
    stage_frag(sWqF, 2, 1, tid, [&](int K, int N) {
        return ((K >> 3) == (N >> 3)) ? Wq[(K >> 3) * 64 + (K & 7) * 8 + (N & 7)] : 0.f;
    });
    stage_frag(sWkF, 2, 1, tid, [&](int K, int N) {
        return ((K >> 3) == (N >> 3)) ? Wk[(K >> 3) * 64 + (K & 7) * 8 + (N & 7)] : 0.f;
    });
    stage_frag(sWvF, 2, 1, tid, [&](int K, int N) {
        return ((K >> 3) == (N >> 3)) ? Wv[(K >> 3) * 64 + (K & 7) * 8 + (N & 7)] : 0.f;
    });
    stage_frag(sWoF, 2, 1, tid, [&](int K, int N) {
        return ((K >> 3) == (N >> 3)) ? Wo[(K >> 3) * 64 + (K & 7) * 8 + (N & 7)] : 0.f;
    });
    stage_frag(sW1F, 2, 1, tid, [&](int K, int N) { return W1[K * 32 + N]; });
    stage_frag(sW2F, 2, 1, tid, [&](int K, int N) { return W2[K * 32 + N]; });

    for (int i = tid; i < 32; i += THREADS) {
        sBmap[i] = (i >= 8) ? b_map[i - 8] : 0.f;
        sBq[i] = bq[i]; sBk[i] = bk[i]; sBv[i] = bv[i]; sBo[i] = bo[i];
        sB1[i] = b1[i]; sB2[i] = b2[i];
        sLg[i] = ln_g[i]; sLb[i] = ln_b[i];
    }
    __syncthreads();

    const int n_  = lane & 15;
    const int hi_ = lane >> 4;

    // ---- Phase 1: image -> patch-permuted f16 tile (16 x 64, zero padded) --
    {
        for (int i = lane; i < 16 * 64; i += 32) w.ast[i] = (_Float16)0.f;
#if USE_TDM
        __builtin_amdgcn_s_wait_tensorcnt(0);       // drain this wave's DMA
        const float4* ip4 = (const float4*)w.u.raw; // ds_load_b128 from LDS
#else
        const float4* ip4 = (const float4*)(img + (size_t)imgIdx * 784);
#endif
        for (int i4 = lane; i4 < 196; i4 += 32) {   // 784 pixels = 196 x float4
            float4 p = ip4[i4];
            int base = i4 * 4;
#pragma unroll
            for (int uu = 0; uu < 4; ++uu) {
                int pix = base + uu;
                int row = pix / 28, col = pix - row * 28;
                int s  = (row / 7) * 4 + (col / 7);
                int pd = (row % 7) * 7 + (col % 7);
                float val = (uu == 0) ? p.x : (uu == 1) ? p.y : (uu == 2) ? p.z : p.w;
                w.ast[s * 64 + pd] = (_Float16)val;
            }
        }
    }
    __syncthreads();

    // ---- Phase 2: x = patches @ Wmap_shifted + bmap_shifted (WMMA) ----
#pragma unroll
    for (int t = 0; t < 2; ++t) {
        v8f c = {};
        c = wmma_f16(load_a16(w.ast, 64, 0,  lane), load_bfrag(sWmapF, t * 2 + 0, lane), c);
        c = wmma_f16(load_a16(w.ast, 64, 32, lane), load_bfrag(sWmapF, t * 2 + 1, lane), c);
        int col = 16 * t + n_;
#pragma unroll
        for (int r = 0; r < 8; ++r)
            w.x[(r + 8 * hi_) * 32 + col] = c[r] + sBmap[col];
    }
    __syncthreads();

    // ---- Phase 3: LayerNorm(x) -> f16 staging ----
    if (lane < 16) {
        float mu = 0.f;
#pragma unroll
        for (int d = 0; d < 32; ++d) mu += w.x[lane * 32 + d];
        mu *= (1.f / 32.f);
        float var = 0.f;
#pragma unroll
        for (int d = 0; d < 32; ++d) { float df = w.x[lane * 32 + d] - mu; var += df * df; }
        var *= (1.f / 32.f);
        float inv = rsqrtf(var + 1e-5f);
#pragma unroll
        for (int d = 0; d < 32; ++d)
            w.ast[lane * 64 + d] = (_Float16)((w.x[lane * 32 + d] - mu) * inv * sLg[d] + sLb[d]);
    }
    __syncthreads();

    // ---- Phase 4: Q/K/V = h @ Wblk + b (block-diagonal heads, WMMA k=32) ----
    {
        const _Float16* Ws[3] = { sWqF, sWkF, sWvF };
        const float*    Bs[3] = { sBq, sBk, sBv };
        _Float16*       Ds[3] = { w.u.q, w.u.k, w.u.v };
#pragma unroll
        for (int p = 0; p < 3; ++p)
#pragma unroll
            for (int t = 0; t < 2; ++t) {
                v8f c = {};
                c = wmma_f16(load_a16(w.ast, 64, 0, lane), load_bfrag(Ws[p], t, lane), c);
                int col = 16 * t + n_;
#pragma unroll
                for (int r = 0; r < 8; ++r)
                    Ds[p][(r + 8 * hi_) * 32 + col] = (_Float16)(c[r] + Bs[p][col]);
            }
    }
    __syncthreads();

    // ---- Phase 5: attention per head (fp32 WMMA 16x16x4) ----
    const float scale = 0.35355339059327373f;  // 1/sqrt(8)
    for (int h = 0; h < NHEAD; ++h) {
        // scores = q_h @ k_h^T   (K = 8 -> two k=4 steps)
        v8f c = {};
#pragma unroll
        for (int ch = 0; ch < 2; ++ch) {
            int e0 = h * 8 + ch * 4 + hi_ * 2;          // even -> 4B aligned pairs
            h2 qa = *(const h2*)(w.u.q + n_ * 32 + e0);
            h2 ka = *(const h2*)(w.u.k + n_ * 32 + e0);
            v2f a, b;
            a.x = (float)qa[0]; a.y = (float)qa[1];
            b.x = (float)ka[0]; b.y = (float)ka[1];
            c = __builtin_amdgcn_wmma_f32_16x16x4_f32(false, a, false, b, (short)0, c, false, false);
        }
#pragma unroll
        for (int r = 0; r < 8; ++r) w.sc[(r + 8 * hi_) * 16 + n_] = c[r] * scale;
        __syncthreads();

        // row softmax (fp32)
        if (lane < 16) {
            float mx = -1e30f;
#pragma unroll
            for (int t = 0; t < 16; ++t) mx = fmaxf(mx, w.sc[lane * 16 + t]);
            float sum = 0.f;
#pragma unroll
            for (int t = 0; t < 16; ++t) {
                float e = __expf(w.sc[lane * 16 + t] - mx);
                w.sc[lane * 16 + t] = e; sum += e;
            }
            float inv = 1.f / sum;
#pragma unroll
            for (int t = 0; t < 16; ++t) w.sc[lane * 16 + t] *= inv;
        }
        __syncthreads();

        // o_h = att @ v_h   (K = 16 -> four k=4 steps; N cols 0..7 valid)
        v8f co = {};
#pragma unroll
        for (int ch = 0; ch < 4; ++ch) {
            int t0 = ch * 4 + hi_ * 2;                  // even -> 8B aligned pair
            v2f a = *(const v2f*)(w.sc + n_ * 16 + t0);
            v2f b;
            b.x = (n_ < 8) ? (float)w.u.v[t0 * 32 + h * 8 + n_] : 0.f;
            b.y = (n_ < 8) ? (float)w.u.v[(t0 + 1) * 32 + h * 8 + n_] : 0.f;
            co = __builtin_amdgcn_wmma_f32_16x16x4_f32(false, a, false, b, (short)0, co, false, false);
        }
        if (n_ < 8) {
#pragma unroll
            for (int r = 0; r < 8; ++r) w.u.o[(r + 8 * hi_) * 32 + h * 8 + n_] = co[r];
        }
        __syncthreads();
    }

    // ---- Phase 6: o -> f16 staging ----
    for (int i = lane; i < 16 * 32; i += 32)
        w.ast[(i >> 5) * 64 + (i & 31)] = (_Float16)w.u.o[i];
    __syncthreads();

    // ---- Phase 7: mhsa = o @ Wo_blk + bo + x  (in-place into x) ----
#pragma unroll
    for (int t = 0; t < 2; ++t) {
        v8f c = {};
        c = wmma_f16(load_a16(w.ast, 64, 0, lane), load_bfrag(sWoF, t, lane), c);
        int col = 16 * t + n_;
#pragma unroll
        for (int r = 0; r < 8; ++r) {
            int idx = (r + 8 * hi_) * 32 + col;
            w.x[idx] = w.x[idx] + c[r] + sBo[col];
        }
    }
    __syncthreads();

    // ---- Phase 8: LayerNorm(mhsa) -> f16 staging ----
    if (lane < 16) {
        float mu = 0.f;
#pragma unroll
        for (int d = 0; d < 32; ++d) mu += w.x[lane * 32 + d];
        mu *= (1.f / 32.f);
        float var = 0.f;
#pragma unroll
        for (int d = 0; d < 32; ++d) { float df = w.x[lane * 32 + d] - mu; var += df * df; }
        var *= (1.f / 32.f);
        float inv = rsqrtf(var + 1e-5f);
#pragma unroll
        for (int d = 0; d < 32; ++d)
            w.ast[lane * 64 + d] = (_Float16)((w.x[lane * 32 + d] - mu) * inv * sLg[d] + sLb[d]);
    }
    __syncthreads();

    // ---- Phase 9: relu(h2 @ W1 + b1) -> f16 (reuse q buffer, stride 32) ----
#pragma unroll
    for (int t = 0; t < 2; ++t) {
        v8f c = {};
        c = wmma_f16(load_a16(w.ast, 64, 0, lane), load_bfrag(sW1F, t, lane), c);
        int col = 16 * t + n_;
#pragma unroll
        for (int r = 0; r < 8; ++r)
            w.u.q[(r + 8 * hi_) * 32 + col] = (_Float16)fmaxf(c[r] + sB1[col], 0.f);
    }
    __syncthreads();

    // ---- Phase 10: enc row0 = (mlp @ W2 + b2 + mhsa)[s=0] -> cls ----
#pragma unroll
    for (int t = 0; t < 2; ++t) {
        v8f c = {};
        c = wmma_f16(load_a16(w.u.q, 32, 0, lane), load_bfrag(sW2F, t, lane), c);
        if (hi_ == 0) {  // row M=0 lives in c[0] of lanes 0..15
            int col = 16 * t + n_;
            w.cls[col] = c[0] + sB2[col] + w.x[col];
        }
    }
    __syncthreads();

    // ---- Phase 11: classifier head (tiny; VALU) ----
    for (int j = lane; j < MLPH; j += 32) {
        float acc = bc1[j];
#pragma unroll
        for (int d = 0; d < 32; ++d) acc += w.cls[d] * Wc1[d * MLPH + j];
        w.hid[j] = fmaxf(acc, 0.f);
    }
    __syncthreads();
    if (lane < NCLS) {
        float acc = bc2[lane];
#pragma unroll
        for (int j = 0; j < MLPH; ++j) acc += w.hid[j] * Wc2[j * NCLS + lane];
        out[(size_t)imgIdx * NCLS + lane] = acc;
    }
}

extern "C" void kernel_launch(void* const* d_in, const int* in_sizes, int n_in,
                              void* d_out, int out_size, void* d_ws, size_t ws_size,
                              hipStream_t stream) {
    (void)n_in; (void)d_ws; (void)ws_size; (void)out_size;
    const float* img   = (const float*)d_in[0];
    const float* W_map = (const float*)d_in[1];
    const float* b_map = (const float*)d_in[2];
    const float* ln_g  = (const float*)d_in[3];
    const float* ln_b  = (const float*)d_in[4];
    const float* Wq = (const float*)d_in[5];  const float* bq = (const float*)d_in[6];
    const float* Wk = (const float*)d_in[7];  const float* bk = (const float*)d_in[8];
    const float* Wv = (const float*)d_in[9];  const float* bv = (const float*)d_in[10];
    const float* Wo = (const float*)d_in[11]; const float* bo = (const float*)d_in[12];
    const float* W1 = (const float*)d_in[13]; const float* b1 = (const float*)d_in[14];
    const float* W2 = (const float*)d_in[15]; const float* b2 = (const float*)d_in[16];
    const float* Wc1 = (const float*)d_in[17]; const float* bc1 = (const float*)d_in[18];
    const float* Wc2 = (const float*)d_in[19]; const float* bc2 = (const float*)d_in[20];

    int nImg = in_sizes[0] / 784;              // B = 32768
    int grid = nImg / WAVES_PER_WG;            // B divisible by 4

    vit_fwd_kernel<<<grid, THREADS, 0, stream>>>(
        img, W_map, b_map, ln_g, ln_b,
        Wq, bq, Wk, bk, Wv, bv, Wo, bo,
        W1, b1, W2, b2, Wc1, bc1, Wc2, bc2,
        (float*)d_out);
}
